// TransformerBlock_31138512896810
// MI455X (gfx1250) — matrix operand order repaired and checked by reading, not compiled, not run on hardware
//
#include <hip/hip_runtime.h>
#include <hip/hip_bf16.h>
#include <math.h>

typedef __attribute__((ext_vector_type(16))) _Float16 v16h;
typedef __attribute__((ext_vector_type(8)))  float    v8f;

#define WMMA_F16(a, b, c) \
    __builtin_amdgcn_wmma_f32_16x16x32_f16(false, (a), false, (b), (short)0, (c), false, false)

union H16 { _Float16 h[16]; uint4 q[2]; };
union H4  { _Float16 h[4];  uint2 u;    };

// ---------------------------------------------------------------------------
// Async global->LDS B128 copy (CDNA5, ASYNCcnt-tracked). Generic LDS pointers
// carry the LDS byte offset in their low 32 bits (aperture mapping) => VDST.
// ---------------------------------------------------------------------------
__device__ __forceinline__ void async_copy_b128(const void* g, void* lds) {
    const uint32_t loff = (uint32_t)(uintptr_t)lds;
    const uint64_t ga   = (uint64_t)(uintptr_t)g;
    asm volatile("global_load_async_to_lds_b128 %0, %1, off"
                 :: "v"(loff), "v"(ga) : "memory");
}
__device__ __forceinline__ void wait_async_zero() {
    asm volatile("s_wait_asynccnt 0x0" ::: "memory");
}

// ---- fragment loaders (CDNA5 wave32 WMMA layouts, f16 16x16x32) -----------
__device__ __forceinline__ v16h load_a_frag(const _Float16* base, int stride, int lane) {
    const int m  = lane & 15;
    const int kh = (lane >> 4) ? 8 : 0;
    const _Float16* row = base + m * stride;
    union { v16h v; uint4 q[2]; } f;
    f.q[0] = *(const uint4*)(row + kh);        // K = kh .. kh+7
    f.q[1] = *(const uint4*)(row + 16 + kh);   // K = 16+kh .. 16+kh+7
    return f.v;
}
__device__ __forceinline__ v16h load_bt_frag(const _Float16* base, int stride, int lane) {
    const int n  = lane & 15;
    const int kb = (lane >> 4) ? 8 : 0;
    const _Float16* row = base + n * stride;
    union { v16h v; uint4 q[2]; } f;
    f.q[0] = *(const uint4*)(row + kb);
    f.q[1] = *(const uint4*)(row + 16 + kb);
    return f.v;
}

// ===========================================================================
// f32 -> f16 elementwise convert (activations). 4 elems / thread.
// ===========================================================================
__global__ __launch_bounds__(256) void cvt16_kernel(_Float16* __restrict__ dst,
                                                    const float* __restrict__ src, int n4) {
    const int i = blockIdx.x * 256 + threadIdx.x;
    if (i >= n4) return;
    const float4 f = ((const float4*)src)[i];
    H4 u;
    u.h[0] = (_Float16)f.x; u.h[1] = (_Float16)f.y;
    u.h[2] = (_Float16)f.z; u.h[3] = (_Float16)f.w;
    ((uint2*)dst)[i] = u.u;
}

// ===========================================================================
// W[K,N] f32  ->  WT[N,K] f16  (64x64 tiles through LDS, coalesced both ways)
// grid: (N/64, K/64), 256 threads.
// ===========================================================================
__global__ __launch_bounds__(256) void wtrans_kernel(_Float16* __restrict__ wt,
                                                     const float* __restrict__ w,
                                                     int K, int N) {
    __shared__ __align__(16) _Float16 st[64 * 72];   // [k_local][n_local]
    const int k0 = blockIdx.y * 64, n0 = blockIdx.x * 64;
    const int tid = threadIdx.x;
    const int r = tid >> 2, c = (tid & 3) * 16;
    {
        const float4* src = (const float4*)(w + (size_t)(k0 + r) * N + n0 + c);
        H16 u;
#pragma unroll
        for (int i = 0; i < 4; ++i) {
            float4 f = src[i];
            u.h[i * 4 + 0] = (_Float16)f.x; u.h[i * 4 + 1] = (_Float16)f.y;
            u.h[i * 4 + 2] = (_Float16)f.z; u.h[i * 4 + 3] = (_Float16)f.w;
        }
        uint4* d = (uint4*)(st + r * 72 + c);
        d[0] = u.q[0]; d[1] = u.q[1];
    }
    __syncthreads();
    H16 o;
#pragma unroll
    for (int e = 0; e < 16; ++e) o.h[e] = st[(c + e) * 72 + r];   // [k=c+e][n=r]
    uint4* dst = (uint4*)(wt + (size_t)(n0 + r) * K + k0 + c);
    dst[0] = o.q[0]; dst[1] = o.q[1];
}

// ===========================================================================
// GEMM: out = act(A16[M,K] @ WT16[N,K]^T + bias), f16 WMMA, async double-buffer.
// 256 threads (8 waves); block tile 128x128; wave tile 32x64; K-step 32.
// Compile-time epilogue: f32 / f16 row-major / f16 per-head-transposed V.
// ===========================================================================
#define GS 40   // 32 halfs + 8 pad: 80B rows, 16B aligned, conflict-free

__device__ __forceinline__ void gemm_stage(const _Float16* __restrict__ A16,
                                           const _Float16* __restrict__ WT16,
                                           _Float16* sA, _Float16* sB,
                                           int m0, int n0, int k0, int K, int tid) {
#pragma unroll
    for (int i = 0; i < 2; ++i) {          // A: 128 rows x 4 chunks(16B) = 512
        const int ch = tid + i * 256;
        const int r = ch >> 2, c = (ch & 3) * 8;
        async_copy_b128(A16 + (size_t)(m0 + r) * K + k0 + c, sA + r * GS + c);
    }
#pragma unroll
    for (int i = 0; i < 2; ++i) {          // B: 128 rows x 4 chunks(16B) = 512
        const int ch = tid + i * 256;
        const int r = ch >> 2, c = (ch & 3) * 8;
        async_copy_b128(WT16 + (size_t)(n0 + r) * K + k0 + c, sB + r * GS + c);
    }
}

template<bool RELU, bool O32, bool O16, bool VT>
__global__ __launch_bounds__(256) void gemm16_kernel(
    const _Float16* __restrict__ A16, const _Float16* __restrict__ WT16,
    const float* __restrict__ bias,
    float* __restrict__ out32, _Float16* __restrict__ out16,
    _Float16* __restrict__ vt16,
    int M, int N, int K, int S)
{
    __shared__ __align__(16) _Float16 sA[2][128 * GS];
    __shared__ __align__(16) _Float16 sB[2][128 * GS];

    const int tid  = threadIdx.x;
    const int lane = tid & 31;
    const int wave = tid >> 5;
    const int wv = wave & 3;       // row block within tile
    const int wh = wave >> 2;      // column half within tile
    const int m0 = blockIdx.y * 128;
    const int n0 = blockIdx.x * 128;

    v8f acc[2][4] = {};

    gemm_stage(A16, WT16, sA[0], sB[0], m0, n0, 0, K, tid);

    const int nsteps = K >> 5;
    for (int s = 0; s < nsteps; ++s) {
        wait_async_zero();            // stage s landed (per-wave ASYNCcnt)
        __syncthreads();              // stage s visible everywhere, buf s^1 free
        if (s + 1 < nsteps)
            gemm_stage(A16, WT16, sA[(s + 1) & 1], sB[(s + 1) & 1],
                       m0, n0, (s + 1) << 5, K, tid);
        if (s + 2 < nsteps) {         // prefetch K+2 into GL2
            const int k2 = (s + 2) << 5;
            __builtin_prefetch(A16 + (size_t)(m0 + (tid >> 1)) * K + k2, 0, 3);
            __builtin_prefetch(WT16 + (size_t)(n0 + (tid >> 1)) * K + k2, 0, 3);
        }
        const _Float16* cA = sA[s & 1];
        const _Float16* cB = sB[s & 1] + wh * 64 * GS;
        v16h bf[4];
#pragma unroll
        for (int nb = 0; nb < 4; ++nb)
            bf[nb] = load_bt_frag(cB + nb * 16 * GS, GS, lane);
#pragma unroll
        for (int mi = 0; mi < 2; ++mi) {
            const v16h a = load_a_frag(cA + (wv * 32 + mi * 16) * GS, GS, lane);
#pragma unroll
            for (int nb = 0; nb < 4; ++nb)
                acc[mi][nb] = WMMA_F16(a, bf[nb], acc[mi][nb]);
        }
    }

    const int hi = (lane >> 4) * 8;
    const int nc = lane & 15;
#pragma unroll
    for (int mi = 0; mi < 2; ++mi) {
#pragma unroll
        for (int nb = 0; nb < 4; ++nb) {
            const int col = n0 + wh * 64 + nb * 16 + nc;
            const float bval = bias[col];
#pragma unroll
            for (int r = 0; r < 8; ++r) {
                const int m = m0 + wv * 32 + mi * 16 + r + hi;
                float v = acc[mi][nb][r] + bval;
                if constexpr (RELU) v = fmaxf(v, 0.0f);
                if constexpr (O32) out32[(size_t)m * N + col] = v;
                if constexpr (O16) out16[(size_t)m * N + col] = (_Float16)v;
                if constexpr (VT) {    // V: per-head transposed [B][H][D][S]
                    const int b2 = m / S, ss = m - b2 * S;
                    const int h = col >> 6, d = col & 63;
                    vt16[(((size_t)b2 * (N >> 6) + h) * 64 + d) * S + ss] = (_Float16)v;
                }
            }
        }
    }
}

// ===========================================================================
// Flash attention, score^{-1/2} + online softmax. f16 in (q,k row-major;
// v pre-transposed [B][H][D][S]), f16 ctx out. grid (S/64, B*H), 128 threads.
// ===========================================================================
#define AS 72

__global__ __launch_bounds__(128) void flash_attn_kernel(
    _Float16* __restrict__ ctx16, const _Float16* __restrict__ Q16,
    const _Float16* __restrict__ K16, const _Float16* __restrict__ VT16,
    const int* __restrict__ maskp, int Bn, int S, int Hn)
{
    const int E = Hn * 64;
    const int qt = blockIdx.x;
    const int bh = blockIdx.y;
    const int b  = bh / Hn, h = bh % Hn;

    __shared__ __align__(16) _Float16 sQ [64 * AS];     // [q_local][d]
    __shared__ __align__(16) _Float16 sK [64 * AS];     // [k_local][d]
    __shared__ __align__(16) _Float16 sVT[64 * AS];     // [d][k_local]
    __shared__ __align__(16) _Float16 sP [4 * 16 * AS]; // per-wave P [q][k_local]

    const int tid  = threadIdx.x;
    const int lane = tid & 31;
    const int wave = tid >> 5;
    const int hi   = (lane >> 4) * 8;
    const int nc   = lane & 15;

    // stage Q once (async): 64 rows x 8 chunks(16B)
#pragma unroll
    for (int i = 0; i < 4; ++i) {
        const int ch = tid + i * 128;
        const int r = ch >> 3, c = (ch & 7) * 8;
        async_copy_b128(Q16 + ((size_t)b * S + qt * 64 + r) * E + h * 64 + c,
                        sQ + r * AS + c);
    }

    float rm[8], rs[8];
#pragma unroll
    for (int r = 0; r < 8; ++r) { rm[r] = -__builtin_inff(); rs[r] = 0.0f; }
    v8f o[4] = {};

    for (int kb = 0; kb < S; kb += 64) {
        __syncthreads();                     // prior compute done with sK/sVT
#pragma unroll
        for (int i = 0; i < 4; ++i) {        // K tile rows
            const int ch = tid + i * 128;
            const int r = ch >> 3, c = (ch & 7) * 8;
            async_copy_b128(K16 + ((size_t)b * S + kb + r) * E + h * 64 + c,
                            sK + r * AS + c);
        }
#pragma unroll
        for (int i = 0; i < 4; ++i) {        // VT tile rows (d-major)
            const int ch = tid + i * 128;
            const int r = ch >> 3, c = (ch & 7) * 8;
            async_copy_b128(VT16 + (((size_t)b * Hn + h) * 64 + r) * S + kb + c,
                            sVT + r * AS + c);
        }
        wait_async_zero();
        __syncthreads();

        // ---- scores: S = Q . K^T --------------------------------------
        v8f sc[4] = {};
#pragma unroll
        for (int kc = 0; kc < 2; ++kc) {
            const v16h a = load_a_frag(sQ + wave * 16 * AS + kc * 32, AS, lane);
#pragma unroll
            for (int nb = 0; nb < 4; ++nb) {
                const v16h bfr = load_bt_frag(sK + nb * 16 * AS + kc * 32, AS, lane);
                sc[nb] = WMMA_F16(a, bfr, sc[nb]);
            }
        }

        // ---- s^{-1/2} transform + online softmax ----------------------
        float p[4][8], bm[8];
#pragma unroll
        for (int r = 0; r < 8; ++r) bm[r] = -__builtin_inff();
#pragma unroll
        for (int nb = 0; nb < 4; ++nb) {
            const int key = kb + nb * 16 + nc;
            const int mv  = maskp[(size_t)b * S + key];
#pragma unroll
            for (int r = 0; r < 8; ++r) {
                const float tv = mv ? rsqrtf(sc[nb][r]) : -__builtin_inff();
                p[nb][r] = tv;
                bm[r] = fmaxf(bm[r], tv);
            }
        }
#pragma unroll
        for (int r = 0; r < 8; ++r) {
#pragma unroll
            for (int off = 1; off < 16; off <<= 1)
                bm[r] = fmaxf(bm[r], __shfl_xor(bm[r], off, 32));
            const float newm  = fmaxf(rm[r], bm[r]);
            const float scale = __expf(rm[r] - newm);
            float part = 0.0f;
#pragma unroll
            for (int nb = 0; nb < 4; ++nb) {
                const float e = __expf(p[nb][r] - newm);
                p[nb][r] = e;
                part += e;
            }
#pragma unroll
            for (int off = 1; off < 16; off <<= 1)
                part += __shfl_xor(part, off, 32);
            rs[r] = rs[r] * scale + part;
            rm[r] = newm;
#pragma unroll
            for (int nb = 0; nb < 4; ++nb) o[nb][r] *= scale;
        }

        // ---- C-layout -> A-layout via per-wave LDS patch ---------------
        _Float16* pw = sP + wave * 16 * AS;
#pragma unroll
        for (int nb = 0; nb < 4; ++nb)
#pragma unroll
            for (int r = 0; r < 8; ++r)
                pw[(r + hi) * AS + nb * 16 + nc] = (_Float16)p[nb][r];

        // ---- O += P . V -----------------------------------------------
#pragma unroll
        for (int kc = 0; kc < 2; ++kc) {
            const v16h a = load_a_frag(pw + kc * 32, AS, lane);
#pragma unroll
            for (int nb = 0; nb < 4; ++nb) {
                const v16h bfr = load_bt_frag(sVT + nb * 16 * AS + kc * 32, AS, lane);
                o[nb] = WMMA_F16(a, bfr, o[nb]);
            }
        }
    }

#pragma unroll
    for (int nb = 0; nb < 4; ++nb) {
#pragma unroll
        for (int r = 0; r < 8; ++r) {
            const size_t q = (size_t)b * S + qt * 64 + wave * 16 + r + hi;
            ctx16[q * E + h * 64 + nb * 16 + nc] = (_Float16)(o[nb][r] / rs[r]);
        }
    }
}

// ===========================================================================
// Fused residual-add + LayerNorm. One block per row; optional f16 copy out.
// ===========================================================================
template<bool O16>
__global__ __launch_bounds__(256) void ln_add_kernel(
    float* __restrict__ out, _Float16* __restrict__ out16,
    const float* __restrict__ a, const float* __restrict__ res,
    const float* __restrict__ g, const float* __restrict__ be, int E)
{
    const int row = blockIdx.x, tid = threadIdx.x;
    const float* pa = a   + (size_t)row * E;
    const float* pb = res + (size_t)row * E;

    float v[4], s = 0.0f, ss = 0.0f;
#pragma unroll
    for (int i = 0; i < 4; ++i) {
        const int idx = tid + i * 256;
        const float x = pa[idx] + pb[idx];
        v[i] = x; s += x; ss += x * x;
    }
#pragma unroll
    for (int off = 16; off > 0; off >>= 1) {
        s  += __shfl_xor(s,  off, 32);
        ss += __shfl_xor(ss, off, 32);
    }
    __shared__ float shs[8], shss[8];
    const int wave = tid >> 5, lane = tid & 31;
    if (lane == 0) { shs[wave] = s; shss[wave] = ss; }
    __syncthreads();
    if (wave == 0) {
        s  = (lane < 8) ? shs[lane]  : 0.0f;
        ss = (lane < 8) ? shss[lane] : 0.0f;
#pragma unroll
        for (int off = 4; off > 0; off >>= 1) {
            s  += __shfl_xor(s,  off, 32);
            ss += __shfl_xor(ss, off, 32);
        }
        if (lane == 0) { shs[0] = s; shss[0] = ss; }
    }
    __syncthreads();
    const float mean = shs[0] / E;
    const float var  = shss[0] / E - mean * mean;
    const float inv  = rsqrtf(var + 1e-5f);
#pragma unroll
    for (int i = 0; i < 4; ++i) {
        const int idx = tid + i * 256;
        const float y = (v[i] - mean) * inv * g[idx] + be[idx];
        out[(size_t)row * E + idx] = y;
        if constexpr (O16) out16[(size_t)row * E + idx] = (_Float16)y;
    }
}

// ===========================================================================
static inline void* carve(char*& p, size_t bytes) {
    void* r = (void*)p;
    p += (bytes + 255) & ~(size_t)255;
    return r;
}

extern "C" void kernel_launch(void* const* d_in, const int* in_sizes, int n_in,
                              void* d_out, int out_size, void* d_ws, size_t ws_size,
                              hipStream_t stream)
{
    (void)in_sizes; (void)n_in; (void)out_size; (void)ws_size;
    constexpr int Bc = 4, Sc = 1024, Ec = 1024, Hc = 16, FF = 4096;
    constexpr int M = Bc * Sc;

    const float* value = (const float*)d_in[0];
    const float* keyi  = (const float*)d_in[1];
    const float* query = (const float*)d_in[2];
    const int*   mask  = (const int*)  d_in[3];
    const float* Wv = (const float*)d_in[4];  const float* bv = (const float*)d_in[5];
    const float* Wk = (const float*)d_in[6];  const float* bk = (const float*)d_in[7];
    const float* Wq = (const float*)d_in[8];  const float* bq = (const float*)d_in[9];
    const float* Wo = (const float*)d_in[10]; const float* bo = (const float*)d_in[11];
    const float* W1 = (const float*)d_in[12]; const float* b1 = (const float*)d_in[13];
    const float* W2 = (const float*)d_in[14]; const float* b2 = (const float*)d_in[15];
    const float* g1  = (const float*)d_in[16]; const float* be1 = (const float*)d_in[17];
    const float* g2  = (const float*)d_in[18]; const float* be2 = (const float*)d_in[19];
    float* out = (float*)d_out;

    char* wp = (char*)d_ws;
    const size_t ME = (size_t)M * Ec;
    _Float16* value16 = (_Float16*)carve(wp, ME * 2);
    _Float16* key16   = (_Float16*)carve(wp, ME * 2);
    _Float16* query16 = (_Float16*)carve(wp, ME * 2);
    _Float16* WvT = (_Float16*)carve(wp, (size_t)Ec * Ec * 2);
    _Float16* WkT = (_Float16*)carve(wp, (size_t)Ec * Ec * 2);
    _Float16* WqT = (_Float16*)carve(wp, (size_t)Ec * Ec * 2);
    _Float16* WoT = (_Float16*)carve(wp, (size_t)Ec * Ec * 2);
    _Float16* W1T = (_Float16*)carve(wp, (size_t)Ec * FF * 2);
    _Float16* W2T = (_Float16*)carve(wp, (size_t)FF * Ec * 2);
    _Float16* vt16  = (_Float16*)carve(wp, ME * 2);       // [B][H][D][S]
    _Float16* kp16  = (_Float16*)carve(wp, ME * 2);
    _Float16* qp16  = (_Float16*)carve(wp, ME * 2);
    _Float16* ctx16 = (_Float16*)carve(wp, ME * 2);
    float*    attn32 = (float*)carve(wp, ME * 4);
    float*    x1_32  = (float*)carve(wp, ME * 4);
    _Float16* x1_16  = (_Float16*)carve(wp, ME * 2);
    _Float16* ff1_16 = (_Float16*)carve(wp, (size_t)M * FF * 2);
    float*    ff2_32 = (float*)carve(wp, ME * 4);

    const dim3 b256(256), b128(128);
    const dim3 gE (Ec / 128, M / 128);
    const dim3 gFF(FF / 128, M / 128);

    // ---- one-time precision/layout passes (bandwidth-trivial) ----------
    const int n4 = (int)(ME / 4);
    cvt16_kernel<<<dim3((n4 + 255) / 256), b256, 0, stream>>>(value16, value, n4);
    cvt16_kernel<<<dim3((n4 + 255) / 256), b256, 0, stream>>>(key16,   keyi,  n4);
    cvt16_kernel<<<dim3((n4 + 255) / 256), b256, 0, stream>>>(query16, query, n4);
    wtrans_kernel<<<dim3(Ec / 64, Ec / 64), b256, 0, stream>>>(WvT, Wv, Ec, Ec);
    wtrans_kernel<<<dim3(Ec / 64, Ec / 64), b256, 0, stream>>>(WkT, Wk, Ec, Ec);
    wtrans_kernel<<<dim3(Ec / 64, Ec / 64), b256, 0, stream>>>(WqT, Wq, Ec, Ec);
    wtrans_kernel<<<dim3(Ec / 64, Ec / 64), b256, 0, stream>>>(WoT, Wo, Ec, Ec);
    wtrans_kernel<<<dim3(FF / 64, Ec / 64), b256, 0, stream>>>(W1T, W1, Ec, FF);
    wtrans_kernel<<<dim3(Ec / 64, FF / 64), b256, 0, stream>>>(W2T, W2, FF, Ec);

    // ---- QKV projections (V straight to per-head-transposed f16) -------
    gemm16_kernel<false, false, false, true><<<gE, b256, 0, stream>>>(
        value16, WvT, bv, nullptr, nullptr, vt16, M, Ec, Ec, Sc);
    gemm16_kernel<false, false, true, false><<<gE, b256, 0, stream>>>(
        key16, WkT, bk, nullptr, kp16, nullptr, M, Ec, Ec, Sc);
    gemm16_kernel<false, false, true, false><<<gE, b256, 0, stream>>>(
        query16, WqT, bq, nullptr, qp16, nullptr, M, Ec, Ec, Sc);
    // ---- flash attention ------------------------------------------------
    flash_attn_kernel<<<dim3(Sc / 64, Bc * Hc), b128, 0, stream>>>(
        ctx16, qp16, kp16, vt16, mask, Bc, Sc, Hc);
    // ---- output projection + LN1 ---------------------------------------
    gemm16_kernel<false, true, false, false><<<gE, b256, 0, stream>>>(
        ctx16, WoT, bo, attn32, nullptr, nullptr, M, Ec, Ec, Sc);
    ln_add_kernel<true><<<dim3(M), b256, 0, stream>>>(x1_32, x1_16, attn32, query, g1, be1, Ec);
    // ---- feed-forward + LN2 --------------------------------------------
    gemm16_kernel<true, false, true, false><<<gFF, b256, 0, stream>>>(
        x1_16, W1T, b1, nullptr, ff1_16, nullptr, M, FF, Ec, Sc);
    gemm16_kernel<false, true, false, false><<<gE, b256, 0, stream>>>(
        ff1_16, W2T, b2, ff2_32, nullptr, nullptr, M, Ec, FF, Sc);
    ln_add_kernel<false><<<dim3(M), b256, 0, stream>>>(out, nullptr, ff2_32, x1_32, g2, be2, Ec);
}
